// EdgeAttention_64115271794804
// MI455X (gfx1250) — matrix operand order repaired and checked by reading, not compiled, not run on hardware
//
#include <hip/hip_runtime.h>

typedef __attribute__((ext_vector_type(16))) _Float16 v16h;
typedef __attribute__((ext_vector_type(8)))  float    v8f;
typedef __attribute__((ext_vector_type(4)))  float    v4f;
typedef __attribute__((ext_vector_type(4)))  unsigned v4u;

#define TILE_M    16
#define DFEAT     128
#define NK_CHUNKS 8     // K = 256 = 8 chunks of 32
#define NT        8     // N = 128 = 8 tiles of 16

// Issue one 32B WMMA B-fragment load (two ds_load_b128) with the fragment
// offset folded into the DS immediate. No wait here: used for pipelining.
template<int OFF>
__device__ inline void frag_load(unsigned base, v4u& lo, v4u& hi) {
    asm volatile("ds_load_b128 %0, %2 offset:%c3\n\t"
                 "ds_load_b128 %1, %2 offset:%c4"
                 : "=v"(lo), "=v"(hi)
                 : "v"(base), "n"(OFF), "n"(OFF + 16));
}

// Wait until at most CNT DS ops outstanding, then materialize the fragment.
// The "+v" tie-through makes every consumer data-dependent on the wait.
template<int CNT>
__device__ inline v16h frag_settle(v4u lo, v4u hi) {
    asm volatile("s_wait_dscnt %c2" : "+v"(lo), "+v"(hi) : "n"(CNT));
    v16h b;
    __builtin_memcpy(&b, &lo, 16);
    __builtin_memcpy((char*)&b + 16, &hi, 16);
    return b;
}

// LDS holds W1 converted to f16 and pre-swizzled into WMMA B-operand fragments:
// fragment f = c*NT + tn ; per lane l: n = l&15, K = c*32 + (l<16 ? 0 : 8) + (i<8 ? i : i+8)
__global__ __launch_bounds__(256)
void edge_attn_wmma_kernel(const float* __restrict__ nf,
                           const unsigned* __restrict__ ei32, // low dwords of int64 edge_index
                           const float* __restrict__ W1,
                           const float* __restrict__ b1,
                           const float* __restrict__ W2,
                           const float* __restrict__ b2,
                           float* __restrict__ out,
                           int nEdges)
{
    __shared__ __align__(32) _Float16 sB[NK_CHUNKS * NT * 512]; // 64 KB

    const int tid = threadIdx.x;

    // ---- one-time per block: convert + swizzle W1 (f32 -> f16 B fragments) ----
    for (int idx = tid; idx < NK_CHUNKS * NT * 512; idx += blockDim.x) {
        int f   = idx >> 9;       // fragment index = c*8 + tn
        int rem = idx & 511;
        int l   = rem >> 4;       // lane within wave
        int i   = rem & 15;       // element within lane
        int c   = f >> 3;
        int tn  = f & 7;
        int n   = l & 15;
        int k   = c * 32 + ((l < 16) ? 0 : 8) + ((i < 8) ? i : i + 8);
        sB[idx] = (_Float16)W1[k * DFEAT + tn * 16 + n];
    }
    __syncthreads();

    const int lid   = tid & 31;
    const int wave  = tid >> 5;
    const int wavesPerBlock = blockDim.x >> 5;
    const int gwave  = blockIdx.x * wavesPerBlock + wave;
    const int nwaves = gridDim.x * wavesPerBlock;
    const int nTiles = (nEdges + TILE_M - 1) / TILE_M;

    const int n15   = lid & 15;
    const int ahalf = (lid < 16) ? 0 : 8;   // ISA 16-bit A 16x32 lane-half K offset

    // base LDS byte offset for this lane's fragment slot
    const unsigned sbLane = (unsigned)(size_t)(const void*)&sB[0] + (unsigned)lid * 32u;

    // epilogue constants: this lane's hidden column within each N tile
    float b1v[NT], w2cv[NT], w2tv[NT];
    #pragma unroll
    for (int tn = 0; tn < NT; ++tn) {
        int col = tn * 16 + n15;
        b1v[tn]  = b1[col];
        w2cv[tn] = W2[col * 2 + 0];
        w2tv[tn] = W2[col * 2 + 1];
    }
    const float b2c = b2[0], b2t = b2[1];

    const v8f vzero = {0.f, 0.f, 0.f, 0.f, 0.f, 0.f, 0.f, 0.f};

    for (int tile = gwave; tile < nTiles; tile += nwaves) {
        const int eBase = tile * TILE_M;
        int eRow = eBase + n15;
        if (eRow >= nEdges) eRow = nEdges - 1;            // clamp (no divergence)
        const unsigned sidx = ei32[2 * (size_t)eRow];                    // int64 low dword
        const unsigned didx = ei32[2 * ((size_t)nEdges + (size_t)eRow)];

        v8f acc[NT];
        #pragma unroll
        for (int tn = 0; tn < NT; ++tn) acc[tn] = vzero;

        #pragma unroll
        for (int c = 0; c < NK_CHUNKS; ++c) {
            // edge_feat row = concat(src_feat, dst_feat); chunk c covers K [c*32, c*32+32)
            const float* rowp = (c < 4)
                ? (nf + (size_t)sidx * DFEAT + c * 32)
                : (nf + (size_t)didx * DFEAT + (c - 4) * 32);
            const v4f* p = (const v4f*)(rowp + ahalf);    // 32B aligned
            v4f x0 = p[0], x1 = p[1];                     // K = ahalf + 0..7
            v4f x2 = p[4], x3 = p[5];                     // K = ahalf + 16..23
            v16h a;
            #pragma unroll
            for (int i = 0; i < 4; ++i) {
                a[i]      = (_Float16)x0[i];
                a[4 + i]  = (_Float16)x1[i];
                a[8 + i]  = (_Float16)x2[i];
                a[12 + i] = (_Float16)x3[i];
            }

            // 8 N-tiles, software-pipelined one fragment (2 ds loads) ahead.
            const unsigned addrc = sbLane + (unsigned)c * (NT * 1024u);
            v4u lo0, hi0, lo1, hi1;
            frag_load<0>(addrc, lo0, hi0);
            #define ATTN_STEP(TN)                                                  \
            {                                                                      \
                frag_load<((TN) + 1) * 1024>(addrc, lo1, hi1);                     \
                v16h bfr = frag_settle<2>(lo0, hi0);                               \
                acc[TN] = __builtin_amdgcn_wmma_f32_16x16x32_f16(                  \
                    false, a, false, bfr, (short)0, acc[TN], false, false);        \
                lo0 = lo1; hi0 = hi1;                                              \
            }
            ATTN_STEP(0) ATTN_STEP(1) ATTN_STEP(2) ATTN_STEP(3)
            ATTN_STEP(4) ATTN_STEP(5) ATTN_STEP(6)
            #undef ATTN_STEP
            {
                v16h bfr = frag_settle<0>(lo0, hi0);
                acc[7] = __builtin_amdgcn_wmma_f32_16x16x32_f16(
                    false, a, false, bfr, (short)0, acc[7], false, false);
            }
        }

        // ---- epilogue: h = relu(acc + b1); scores = h @ W2; reduce over N ----
        float sc[8], st[8];
        #pragma unroll
        for (int r = 0; r < 8; ++r) { sc[r] = 0.f; st[r] = 0.f; }
        #pragma unroll
        for (int tn = 0; tn < NT; ++tn) {
            #pragma unroll
            for (int r = 0; r < 8; ++r) {
                float h = acc[tn][r] + b1v[tn];
                h = fmaxf(h, 0.0f);
                sc[r] += h * w2cv[tn];
                st[r] += h * w2tv[tn];
            }
        }
        // reduce across the 16 lanes spanning the hidden (N) dimension
        #pragma unroll
        for (int r = 0; r < 8; ++r) {
            #pragma unroll
            for (int m = 1; m <= 8; m <<= 1) {
                sc[r] += __shfl_xor(sc[r], m, 32);
                st[r] += __shfl_xor(st[r], m, 32);
            }
        }
        // lane 0 writes rows 0..7, lane 16 writes rows 8..15 (C layout M split)
        if (n15 == 0) {
            const int mbase = (lid >> 4) * 8;
            #pragma unroll
            for (int r = 0; r < 8; ++r) {
                int e = eBase + mbase + r;
                if (e < nEdges) {
                    float s0 = sc[r] + b2c;
                    float s1 = st[r] + b2t;
                    float mx  = fmaxf(s0, s1);
                    float e0  = __expf(s0 - mx);
                    float e1  = __expf(s1 - mx);
                    float inv = 1.0f / (e0 + e1);
                    out[e]                  = e0 * inv;  // edge_c
                    out[(size_t)nEdges + e] = e1 * inv;  // edge_t
                }
            }
        }
    }
}

extern "C" void kernel_launch(void* const* d_in, const int* in_sizes, int n_in,
                              void* d_out, int out_size, void* d_ws, size_t ws_size,
                              hipStream_t stream) {
    const float*    nf   = (const float*)d_in[0];
    const unsigned* ei32 = (const unsigned*)d_in[1];   // int64 (2,E): use low dwords
    const float*    W1   = (const float*)d_in[2];
    const float*    b1   = (const float*)d_in[3];
    const float*    W2   = (const float*)d_in[4];
    const float*    b2   = (const float*)d_in[5];
    float*          out  = (float*)d_out;

    const int nEdges = in_sizes[1] / 2;
    const int nTiles = (nEdges + TILE_M - 1) / TILE_M;

    int blocks = 1024;                                  // grid-stride over tiles
    int needed = (nTiles + 7) / 8;                      // 8 waves per block
    if (blocks > needed) blocks = needed;
    if (blocks < 1) blocks = 1;

    edge_attn_wmma_kernel<<<blocks, 256, 0, stream>>>(nf, ei32, W1, b1, W2, b2,
                                                      out, nEdges);
}
